// QWenAttention_88553635708933
// MI455X (gfx1250) — compile-verified, matrix operand order checked by reading, not run on hardware
//
#include <hip/hip_runtime.h>
#include <hip/hip_bf16.h>
#include <math.h>

#define B_  2
#define S_  2048
#define H_  4096
#define NH_ 32
#define HD_ 128
#define BS_ (B_ * S_)            // 4096 token rows

typedef __attribute__((ext_vector_type(16))) __bf16 v16bf;
typedef __attribute__((ext_vector_type(8)))  float  v8f;

union Frag16 { v16bf v; unsigned short u[16]; uint4 q[2]; };

__device__ __forceinline__ unsigned short f2bf(float f) {
    unsigned int u = __float_as_uint(f);
    u += 0x7FFFu + ((u >> 16) & 1u);       // round-to-nearest-even
    return (unsigned short)(u >> 16);
}

// Swizzled 32x16 fragment tile: element (kk in 0..31, nn in 0..15) lives at
// lane = nn + 16*((kk>>3)&1), idx = 8*((kk>>4)&1) + (kk&7)  -> tile[lane*16+idx]
// Matches the CDNA5 16-bit B-matrix VGPR layout: a lane's 16 values are 32
// contiguous bytes (=> two b128 loads).
__device__ __forceinline__ int swz_idx(int kk, int nn) {
    int h  = (kk >> 3) & 1;
    int hi = (kk >> 4) & 1;
    return (nn + 16 * h) * 16 + hi * 8 + (kk & 7);
}

__device__ __forceinline__ v16bf load_frag_swz(const unsigned short* tile, int lane) {
    Frag16 f;
    const uint4* p = (const uint4*)(tile + lane * 16);
    f.q[0] = p[0];
    f.q[1] = p[1];
    return f.v;
}

// A (16x32, MxK) from row-major memory: per-lane data is two contiguous
// 16-byte runs -> two b128 loads.
__device__ __forceinline__ v16bf load_frag_a(const unsigned short* src, int ld, int lane) {
    int m = lane & 15, half = lane >> 4;
    Frag16 f;
    const uint4* p0 = (const uint4*)(src + (size_t)m * ld + 8 * half);
    const uint4* p1 = (const uint4*)(src + (size_t)m * ld + 16 + 8 * half);
    f.q[0] = p0[0];
    f.q[1] = p1[0];
    return f.v;
}

// ---- f32 -> bf16 (row-major) -----------------------------------------------
__global__ __launch_bounds__(256) void cvt_bf16_kernel(const float* __restrict__ src,
                                                       unsigned short* __restrict__ dst,
                                                       size_t n) {
    size_t i = (size_t)blockIdx.x * 256 + threadIdx.x;
    if (i < n) dst[i] = f2bf(src[i]);
}

// ---- f32 row-major [K][N] -> bf16 swizzled fragment tiles ------------------
__global__ __launch_bounds__(256) void swz_weight_kernel(const float* __restrict__ src,
                                                         unsigned short* __restrict__ dst,
                                                         int K, int N) {
    size_t i = (size_t)blockIdx.x * 256 + threadIdx.x;
    if (i >= (size_t)K * N) return;
    int n = (int)(i % N);
    int k = (int)(i / N);
    size_t tile = (size_t)(k >> 5) * (N >> 4) + (n >> 4);
    dst[tile * 512 + swz_idx(k & 31, n & 15)] = f2bf(src[i]);
}

// ---- Tiled bf16 GEMM: C(f32,MxN) = A(bf16 row-major) * B(bf16 swizzled) ----
// Block 256 threads = 8 waves; tile 128x128x32; wave -> 32x64 (2x4 frags).
// Double-buffered LDS; N/K compile-time so all addressing folds to immediates.
template <int N, int K, bool HAS_BIAS>
__global__ __launch_bounds__(256)
void gemm_bf16_kernel(const unsigned short* __restrict__ A,
                      const unsigned short* __restrict__ Bswz,
                      const float* __restrict__ bias,
                      float* __restrict__ C) {
    constexpr int NT16   = N >> 4;        // swizzled tiles per K-slice
    constexpr int KSTEPS = K / 32;
    __shared__ __align__(16) unsigned short As[2][128 * 40];
    __shared__ __align__(16) unsigned short Bs[2][8 * 512];

    const int tid  = threadIdx.x;
    const int lane = tid & 31;
    const int wid  = tid >> 5;
    const int wm   = wid & 3;             // 4 waves over M
    const int wn   = wid >> 2;            // 2 waves over N
    const int bm   = blockIdx.y;
    const int bn   = blockIdx.x;

    const int arow = tid >> 1, aseg = tid & 1;   // A: 128 rows x 2 segs of 16
    const unsigned short* Ag = A + (size_t)(bm * 128 + arow) * K + aseg * 16;
    const unsigned short* Bg = Bswz + (size_t)bn * 8 * 512 + tid * 16;

    v8f acc[2][4] = {};

    // stage 0
    uint4 ra0 = ((const uint4*)Ag)[0];
    uint4 ra1 = ((const uint4*)Ag)[1];
    uint4 rb0 = ((const uint4*)Bg)[0];
    uint4 rb1 = ((const uint4*)Bg)[1];
    uint4 rb2 = ((const uint4*)(Bg + 4096))[0];
    uint4 rb3 = ((const uint4*)(Bg + 4096))[1];
    {
        uint4* da = (uint4*)(As[0] + arow * 40 + aseg * 16);
        da[0] = ra0; da[1] = ra1;
        uint4* db = (uint4*)(Bs[0] + tid * 16);
        db[0] = rb0; db[1] = rb1;
        db = (uint4*)(Bs[0] + 4096 + tid * 16);
        db[0] = rb2; db[1] = rb3;
    }
    __syncthreads();

    for (int kt = 0; kt < KSTEPS; ++kt) {
        const int cur = kt & 1, nxt = cur ^ 1;
        if (kt + 1 < KSTEPS) {           // issue next tile's global loads early
            const unsigned short* Agn = Ag + (kt + 1) * 32;
            const unsigned short* Bgn = Bg + (size_t)(kt + 1) * NT16 * 512;
            ra0 = ((const uint4*)Agn)[0];
            ra1 = ((const uint4*)Agn)[1];
            rb0 = ((const uint4*)Bgn)[0];
            rb1 = ((const uint4*)Bgn)[1];
            rb2 = ((const uint4*)(Bgn + 4096))[0];
            rb3 = ((const uint4*)(Bgn + 4096))[1];
            __builtin_prefetch(Agn + 32, 0, 0);
            __builtin_prefetch(Bgn + (size_t)NT16 * 512, 0, 0);
        }

        v16bf af0 = load_frag_a(As[cur] + (wm * 32)      * 40, 40, lane);
        v16bf af1 = load_frag_a(As[cur] + (wm * 32 + 16) * 40, 40, lane);
#pragma unroll
        for (int jn = 0; jn < 4; ++jn) {
            v16bf bfr = load_frag_swz(Bs[cur] + (wn * 4 + jn) * 512, lane);
            acc[0][jn] = __builtin_amdgcn_wmma_f32_16x16x32_bf16(
                false, af0, false, bfr, (short)0, acc[0][jn], false, false);
            acc[1][jn] = __builtin_amdgcn_wmma_f32_16x16x32_bf16(
                false, af1, false, bfr, (short)0, acc[1][jn], false, false);
        }

        if (kt + 1 < KSTEPS) {           // stage into the other buffer
            uint4* da = (uint4*)(As[nxt] + arow * 40 + aseg * 16);
            da[0] = ra0; da[1] = ra1;
            uint4* db = (uint4*)(Bs[nxt] + tid * 16);
            db[0] = rb0; db[1] = rb1;
            db = (uint4*)(Bs[nxt] + 4096 + tid * 16);
            db[0] = rb2; db[1] = rb3;
        }
        __syncthreads();
    }

    // epilogue: all offsets compile-time; bias loaded once per column block
    const int n  = lane & 15;
    const int mh = (lane >> 4) * 8;       // D-layout: upper lanes hold M=8..15
#pragma unroll
    for (int jn = 0; jn < 4; ++jn) {
        const int col = bn * 128 + wn * 64 + jn * 16 + n;
        float bv = 0.0f;
        if constexpr (HAS_BIAS) bv = bias[col];
#pragma unroll
        for (int im = 0; im < 2; ++im) {
            const int row0 = bm * 128 + wm * 32 + im * 16 + mh;
            float* Cp = C + (size_t)row0 * N + col;
#pragma unroll
            for (int r = 0; r < 8; ++r)
                Cp[r * N] = acc[im][jn][r] + bv;
        }
    }
}

// ---- RoPE (NeoX) + bf16 cast + fragment-swizzled K^T / V -------------------
// Q  -> bf16 row-major [b,h,s,d], pre-scaled by HD^-0.5
// Kt -> swizzled tiles of K^T per head: tile (d/32)*(S/16) + (s/16)
// V  -> swizzled tiles of V   per head: tile (s/32)*(HD/16) + (d/16)
__global__ __launch_bounds__(256)
void rope_kernel(const int* __restrict__ positions, const float* __restrict__ qkv,
                 unsigned short* __restrict__ Q, unsigned short* __restrict__ Kt,
                 unsigned short* __restrict__ V) {
    size_t i = (size_t)blockIdx.x * 256 + threadIdx.x;   // B*S*NH*64 = 2^23
    if (i >= (size_t)B_ * S_ * NH_ * 64) return;
    int d = (int)(i & 63);
    int h = (int)((i >> 6) & 31);
    int s = (int)((i >> 11) & 2047);
    int b = (int)(i >> 22);

    int pos = positions[b * S_ + s];
    float inv = __expf((float)d * (-9.2103403719761840f / 64.0f)); // theta^(-d/64)
    float fr  = (float)pos * inv;
    float sn, cs;
    __sincosf(fr, &sn, &cs);

    const float* base = qkv + (size_t)(b * S_ + s) * (3 * H_);
    float q1 = base[h * HD_ + d],          q2 = base[h * HD_ + d + 64];
    float k1 = base[H_ + h * HD_ + d],     k2 = base[H_ + h * HD_ + d + 64];
    float v1 = base[2 * H_ + h * HD_ + d], v2 = base[2 * H_ + h * HD_ + d + 64];

    const float SC = 0.08838834764831845f;   // HD^-0.5 folded into Q
    size_t head = (size_t)(b * NH_ + h);

    size_t qidx = (head * S_ + s) * HD_ + d;
    Q[qidx]      = f2bf((q1 * cs - q2 * sn) * SC);
    Q[qidx + 64] = f2bf((q2 * cs + q1 * sn) * SC);

    float kr1 = k1 * cs - k2 * sn;
    float kr2 = k2 * cs + k1 * sn;
    unsigned short* Kh = Kt + head * (size_t)HD_ * S_;
    {   // element (kk = d, nn = s) of K^T
        int d2 = d;
        Kh[((size_t)(d2 >> 5) * (S_ >> 4) + (s >> 4)) * 512 + swz_idx(d2 & 31, s & 15)] = f2bf(kr1);
        d2 = d + 64;
        Kh[((size_t)(d2 >> 5) * (S_ >> 4) + (s >> 4)) * 512 + swz_idx(d2 & 31, s & 15)] = f2bf(kr2);
    }
    unsigned short* Vh = V + head * (size_t)S_ * HD_;
    {   // element (kk = s, nn = d) of V
        int d2 = d;
        Vh[((size_t)(s >> 5) * (HD_ >> 4) + (d2 >> 4)) * 512 + swz_idx(s & 31, d2 & 15)] = f2bf(v1);
        d2 = d + 64;
        Vh[((size_t)(s >> 5) * (HD_ >> 4) + (d2 >> 4)) * 512 + swz_idx(s & 31, d2 & 15)] = f2bf(v2);
    }
}

// ---- Flash attention: 1 wave per 16-query tile, 32-key steps ---------------
__global__ __launch_bounds__(128)
void fattn_kernel(const unsigned short* __restrict__ Q,
                  const unsigned short* __restrict__ Kt,
                  const unsigned short* __restrict__ V,
                  unsigned short* __restrict__ attn) {
    __shared__ __align__(16) unsigned short Pt[4][512];  // per-wave P as A-frag tile

    const int tid  = threadIdx.x;
    const int lane = tid & 31;
    const int w    = tid >> 5;
    const int bh   = blockIdx.x >> 5;           // 32 blocks of 4 q-tiles per head
    const int grp  = blockIdx.x & 31;
    const int b    = bh >> 5;                   // NH_ = 32
    const int h    = bh & 31;
    const int q0   = (grp * 4 + w) * 16;

    const size_t head = (size_t)(b * NH_ + h);
    const unsigned short* Qb  = Q  + (head * S_ + q0) * HD_;
    const unsigned short* Ktb = Kt + head * (size_t)HD_ * S_;
    const unsigned short* Vb  = V  + head * (size_t)S_ * HD_;

    v16bf qa[4];
#pragma unroll
    for (int c = 0; c < 4; ++c) qa[c] = load_frag_a(Qb + c * 32, HD_, lane);

    v8f o[8] = {};
    float mrun[8], lrun[8];
#pragma unroll
    for (int r = 0; r < 8; ++r) { mrun[r] = -3.0e38f; lrun[r] = 0.0f; }

    const int n  = lane & 15;
    const int mh = (lane >> 4) * 8;
    const int hN = (n >> 3) & 1;      // swizzle half for P stores
    const int rr = n & 7;
    const int ntiles = (q0 + 16 + 31) >> 5;

    for (int kt = 0; kt < ntiles; ++kt) {
        const int k0 = kt * 32;
        v8f s0 = {}, s1 = {};
#pragma unroll
        for (int c = 0; c < 4; ++c) {
            const unsigned short* Kt0 =
                Ktb + ((size_t)c * (S_ >> 4) + (k0 >> 4)) * 512;
            v16bf kb0 = load_frag_swz(Kt0, lane);
            s0 = __builtin_amdgcn_wmma_f32_16x16x32_bf16(false, qa[c], false, kb0,
                                                         (short)0, s0, false, false);
            v16bf kb1 = load_frag_swz(Kt0 + 512, lane);
            s1 = __builtin_amdgcn_wmma_f32_16x16x32_bf16(false, qa[c], false, kb1,
                                                         (short)0, s1, false, false);
        }
        // online softmax (rows live in 16-lane halves; xor-reduce over n)
#pragma unroll
        for (int r = 0; r < 8; ++r) {
            int q = q0 + mh + r;
            float a = s0[r]; if (k0 + n      > q) a = -3.0e38f;
            float c = s1[r]; if (k0 + 16 + n > q) c = -3.0e38f;
            float rm = fmaxf(a, c);
#pragma unroll
            for (int x = 1; x < 16; x <<= 1) rm = fmaxf(rm, __shfl_xor(rm, x, 32));
            float mnew  = fmaxf(mrun[r], rm);
            float scale = __expf(mrun[r] - mnew);
            float p0 = __expf(a - mnew);
            float p1 = __expf(c - mnew);
            float rs = p0 + p1;
#pragma unroll
            for (int x = 1; x < 16; x <<= 1) rs += __shfl_xor(rs, x, 32);
            lrun[r] = lrun[r] * scale + rs;
            mrun[r] = mnew;
#pragma unroll
            for (int f = 0; f < 8; ++f) o[f][r] *= scale;
            // store P directly in A-fragment swizzle: row m=mh+r, kcol=n / 16+n
            Pt[w][(mh + r + 16 * hN) * 16 + rr]     = f2bf(p0);
            Pt[w][(mh + r + 16 * hN) * 16 + 8 + rr] = f2bf(p1);
        }
        asm volatile("s_wait_dscnt 0" ::: "memory");   // wave-local LDS RAW fence
        v16bf pa = load_frag_swz(&Pt[w][0], lane);
#pragma unroll
        for (int f = 0; f < 8; ++f) {
            v16bf vb = load_frag_swz(Vb + ((size_t)(k0 >> 5) * (HD_ >> 4) + f) * 512, lane);
            o[f] = __builtin_amdgcn_wmma_f32_16x16x32_bf16(false, pa, false, vb,
                                                           (short)0, o[f], false, false);
        }
    }
    // normalize and emit bf16 attn [b, s, h, d] (row-major A for the out-proj)
    unsigned short* ab = attn + ((size_t)b * S_ + q0 + mh) * (NH_ * HD_) + h * HD_ + n;
#pragma unroll
    for (int r = 0; r < 8; ++r) {
        float inv = 1.0f / lrun[r];
#pragma unroll
        for (int f = 0; f < 8; ++f)
            ab[(size_t)r * (NH_ * HD_) + f * 16] = f2bf(o[f][r] * inv);
    }
}

// ---- host orchestration ----------------------------------------------------
extern "C" void kernel_launch(void* const* d_in, const int* in_sizes, int n_in,
                              void* d_out, int out_size, void* d_ws, size_t ws_size,
                              hipStream_t stream) {
    const int*   positions = (const int*)d_in[0];
    const float* hidden    = (const float*)d_in[1];
    const float* W_qkv     = (const float*)d_in[2];
    const float* b_qkv     = (const float*)d_in[3];
    const float* W_o       = (const float*)d_in[4];
    float*       out       = (float*)d_out;

    const size_t XBF_N  = (size_t)BS_ * H_;            // 16M
    const size_t WQKV_N = (size_t)H_ * 3 * H_;         // 50M
    const size_t WO_N   = (size_t)H_ * H_;             // 16M
    const size_t QKV_N  = (size_t)BS_ * 3 * H_;        // 50M
    const size_t HEAD_N = (size_t)B_ * NH_ * S_ * HD_; // 16M

    char* p = (char*)d_ws;
    auto carve = [&](size_t bytes) -> char* {
        char* r = p;
        p += (bytes + 255) & ~(size_t)255;
        return r;
    };
    unsigned short* Xbf    = (unsigned short*)carve(XBF_N * 2);
    unsigned short* Wqkvbf = (unsigned short*)carve(WQKV_N * 2);
    unsigned short* Wobf   = (unsigned short*)carve(WO_N * 2);
    float*          qkvf   = (float*)carve(QKV_N * 4);
    unsigned short* Qbf    = (unsigned short*)carve(HEAD_N * 2);
    unsigned short* Ktbf   = (unsigned short*)carve(HEAD_N * 2);
    unsigned short* Vbf    = (unsigned short*)carve(HEAD_N * 2);
    unsigned short* Attnbf = (unsigned short*)carve(XBF_N * 2);

    cvt_bf16_kernel<<<(unsigned)((XBF_N + 255) / 256), 256, 0, stream>>>(hidden, Xbf, XBF_N);
    swz_weight_kernel<<<(unsigned)((WQKV_N + 255) / 256), 256, 0, stream>>>(W_qkv, Wqkvbf, H_, 3 * H_);
    swz_weight_kernel<<<(unsigned)((WO_N + 255) / 256), 256, 0, stream>>>(W_o, Wobf, H_, H_);

    // QKV projection: [4096 x 4096] x [4096 x 12288] + bias
    gemm_bf16_kernel<3 * H_, H_, true><<<dim3(3 * H_ / 128, BS_ / 128), 256, 0, stream>>>(
        Xbf, Wqkvbf, b_qkv, qkvf);

    rope_kernel<<<(unsigned)(((size_t)B_ * S_ * NH_ * 64 + 255) / 256), 256, 0, stream>>>(
        positions, qkvf, Qbf, Ktbf, Vbf);

    // flash attention: B*NH heads, 32 blocks (of 4 wave-tiles) per head
    fattn_kernel<<<B_ * NH_ * 32, 128, 0, stream>>>(Qbf, Ktbf, Vbf, Attnbf);

    // output projection: [4096 x 4096] x [4096 x 4096]
    gemm_bf16_kernel<H_, H_, false><<<dim3(H_ / 128, BS_ / 128), 256, 0, stream>>>(
        Attnbf, Wobf, nullptr, out);
}
